// UniCrystalFormer_28527172780860
// MI455X (gfx1250) — matrix operand order repaired and checked by reading, not compiled, not run on hardware
//
#include <hip/hip_runtime.h>

// ---------------- problem constants ----------------
#define NNODE 4096
#define NEDGE 32768
#define NGRAPH 128
#define HIDC 128
#define NHEADS 4
#define EHROWS (NEDGE * NHEADS)   // 131072

typedef __attribute__((ext_vector_type(16))) _Float16 v16h;
typedef __attribute__((ext_vector_type(8)))  _Float16 v8h;
typedef __attribute__((ext_vector_type(8)))  float    v8f;

__device__ __forceinline__ float fast_sigmoid(float v) {
  return __builtin_amdgcn_rcpf(1.f + __expf(-v));   // v_rcp_f32, not IEEE divide
}
__device__ __forceinline__ float actf(float v, int act) {
  if (act == 1) return v * fast_sigmoid(v);         // silu
  if (act == 2) return fast_sigmoid(v);             // sigmoid
  return v;
}

// =====================================================================
// WMMA GEMM: C[M,N] = act(A[M,K] @ W[K,N] + bias[N])
// REQUIRES: M%128==0, N%128==0, K%32==0 (true for all model GEMMs except
// megnet_proj which uses k_lin_small).
// 256 threads = 8 wave32 waves; block tile 128x128; K-step 32.
// Wave grid 4x2; each wave computes 32x64 => 8 accumulators, 8 WMMA/K-step.
// A staged row-major in LDS (b128 aligned), B staged transposed so both
// fragment types are contiguous v8h (ds_load_b128) reads.
// ACT is a template parameter so the epilogue is branch-free.
// =====================================================================
template <int ACT>
__global__ __launch_bounds__(256) void k_gemm(const float* __restrict__ A,
                                              const float* __restrict__ Wm,
                                              const float* __restrict__ bias,
                                              float* __restrict__ C,
                                              int M, int N, int K) {
  __shared__ alignas(16) _Float16 As[128][40];    // [m][k]
  __shared__ alignas(16) _Float16 BsT[128][40];   // [n][k]  (transposed)
  const int m0 = blockIdx.x * 128, n0 = blockIdx.y * 128;
  const int t = threadIdx.x;
  const int lane = t & 31, wave = t >> 5;
  const int wr = wave & 3, wc = wave >> 2;   // 4x2 wave grid
  const int lm = lane & 15, lh = lane >> 4;

  v8f acc[8];
  const v8f vz = {0.f, 0.f, 0.f, 0.f, 0.f, 0.f, 0.f, 0.f};
#pragma unroll
  for (int i = 0; i < 8; ++i) acc[i] = vz;

  // A fill mapping: thread -> (row = t>>1, c0 = (t&1)*16): 16 halfs via 4x float4
  const int a_r = t >> 1, a_c0 = (t & 1) * 16;
  // B fill mapping: thread -> (n = t>>1, kb = (t&1)*16): 16 halfs from W column
  const int b_n = t >> 1, b_kb = (t & 1) * 16;

  for (int kt = 0; kt < K; kt += 32) {
    {
      const float4* ap = (const float4*)(A + (size_t)(m0 + a_r) * K + kt + a_c0);
      float4 f0 = ap[0], f1 = ap[1], f2 = ap[2], f3 = ap[3];
      v8h p0, p1;
      p0[0] = (_Float16)f0.x; p0[1] = (_Float16)f0.y; p0[2] = (_Float16)f0.z; p0[3] = (_Float16)f0.w;
      p0[4] = (_Float16)f1.x; p0[5] = (_Float16)f1.y; p0[6] = (_Float16)f1.z; p0[7] = (_Float16)f1.w;
      p1[0] = (_Float16)f2.x; p1[1] = (_Float16)f2.y; p1[2] = (_Float16)f2.z; p1[3] = (_Float16)f2.w;
      p1[4] = (_Float16)f3.x; p1[5] = (_Float16)f3.y; p1[6] = (_Float16)f3.z; p1[7] = (_Float16)f3.w;
      *(v8h*)&As[a_r][a_c0]     = p0;
      *(v8h*)&As[a_r][a_c0 + 8] = p1;
    }
    {
      const float* wp = Wm + (size_t)(kt + b_kb) * N + n0 + b_n;
      v8h p0, p1;
#pragma unroll
      for (int j = 0; j < 8; ++j) p0[j] = (_Float16)wp[(size_t)j * N];
#pragma unroll
      for (int j = 0; j < 8; ++j) p1[j] = (_Float16)wp[(size_t)(j + 8) * N];
      *(v8h*)&BsT[b_n][b_kb]     = p0;
      *(v8h*)&BsT[b_n][b_kb + 8] = p1;
    }
    __syncthreads();

    // A fragments for this wave's two 16-row blocks
    const int kb = lh * 8;
    const int row0 = wr * 32 + lm;
    v8h a0lo = *(const v8h*)&As[row0][kb];
    v8h a0hi = *(const v8h*)&As[row0][kb + 16];
    v8h a1lo = *(const v8h*)&As[row0 + 16][kb];
    v8h a1hi = *(const v8h*)&As[row0 + 16][kb + 16];
    v16h a0 = __builtin_shufflevector(a0lo, a0hi, 0, 1, 2, 3, 4, 5, 6, 7, 8, 9, 10, 11, 12, 13, 14, 15);
    v16h a1 = __builtin_shufflevector(a1lo, a1hi, 0, 1, 2, 3, 4, 5, 6, 7, 8, 9, 10, 11, 12, 13, 14, 15);

#pragma unroll
    for (int nb = 0; nb < 4; ++nb) {
      const int col = wc * 64 + nb * 16 + lm;
      v8h blo = *(const v8h*)&BsT[col][lh * 8];
      v8h bhi = *(const v8h*)&BsT[col][lh * 8 + 16];
      v16h bf = __builtin_shufflevector(blo, bhi, 0, 1, 2, 3, 4, 5, 6, 7, 8, 9, 10, 11, 12, 13, 14, 15);
      acc[nb * 2 + 0] = __builtin_amdgcn_wmma_f32_16x16x32_f16(false, a0, false, bf, (short)0,
                                                               acc[nb * 2 + 0], false, false);
      acc[nb * 2 + 1] = __builtin_amdgcn_wmma_f32_16x16x32_f16(false, a1, false, bf, (short)0,
                                                               acc[nb * 2 + 1], false, false);
    }
    __syncthreads();
  }

#pragma unroll
  for (int nb = 0; nb < 4; ++nb) {
    const int n = n0 + wc * 64 + nb * 16 + lm;
    const float bi = bias[n];
#pragma unroll
    for (int ar = 0; ar < 2; ++ar) {
      v8f ca = acc[nb * 2 + ar];
#pragma unroll
      for (int r = 0; r < 8; ++r) {
        int m = m0 + wr * 32 + ar * 16 + lh * 8 + r;
        C[(size_t)m * N + n] = actf(ca[r] + bi, ACT);
      }
    }
  }
}

// small-K fallback (megnet_proj: M=4096, K=16, N=128)
__global__ void k_lin_small(const float* __restrict__ A, const float* __restrict__ W,
                            const float* __restrict__ b, float* __restrict__ C,
                            int M, int N, int K, int act) {
  long long i = (long long)blockIdx.x * 256 + threadIdx.x;
  if (i >= (long long)M * N) return;
  int r = (int)(i / N), c = (int)(i % N);
  float s = b[c];
  for (int k = 0; k < K; ++k) s += A[(size_t)r * K + k] * W[(size_t)k * N + c];
  C[i] = actf(s, act);
}

// ---------------- elementwise / segment kernels ----------------
__global__ void k_zero(float* p, long long n) {
  long long i = (long long)blockIdx.x * 256 + threadIdx.x;
  if (i < n) p[i] = 0.f;
}

__global__ void k_gather_rows(const float* __restrict__ tab, const int* __restrict__ idx,
                              float* __restrict__ out, int M, int D) {
  long long i = (long long)blockIdx.x * 256 + threadIdx.x;
  if (i >= (long long)M * D) return;
  int r = (int)(i / D), c = (int)(i % D);
  out[i] = tab[(size_t)idx[r] * D + c];
}

__global__ void k_concat2(const float* __restrict__ A, int Da, const float* __restrict__ B, int Db,
                          float* __restrict__ C, int M) {
  int D = Da + Db;
  long long i = (long long)blockIdx.x * 256 + threadIdx.x;
  if (i >= (long long)M * D) return;
  int r = (int)(i / D), c = (int)(i % D);
  C[i] = (c < Da) ? A[(size_t)r * Da + c] : B[(size_t)r * Db + (c - Da)];
}

__global__ void k_gate_fuse(const float* __restrict__ g, const float* __restrict__ e,
                            const float* __restrict__ m, float* __restrict__ out) {
  long long i = (long long)blockIdx.x * 256 + threadIdx.x;
  if (i >= (long long)NNODE * HIDC) return;
  float gv = g[i];
  out[i] = gv * e[i] + (1.f - gv) * m[i];
}

__global__ __launch_bounds__(128) void k_layernorm(const float* __restrict__ x,
                                                   const float* __restrict__ g,
                                                   const float* __restrict__ b,
                                                   float* __restrict__ out, int D) {
  int row = blockIdx.x, t = threadIdx.x;
  const float* xr = x + (size_t)row * D;
  float* orow = out + (size_t)row * D;
  __shared__ float red[128]; __shared__ float s_m, s_r;
  float s = 0.f;
  for (int c = t; c < D; c += 128) s += xr[c];
  red[t] = s; __syncthreads();
  for (int o = 64; o > 0; o >>= 1) { if (t < o) red[t] += red[t + o]; __syncthreads(); }
  if (t == 0) s_m = red[0] / D; __syncthreads();
  float m = s_m, v = 0.f;
  for (int c = t; c < D; c += 128) { float d2 = xr[c] - m; v += d2 * d2; }
  red[t] = v; __syncthreads();
  for (int o = 64; o > 0; o >>= 1) { if (t < o) red[t] += red[t + o]; __syncthreads(); }
  if (t == 0) s_r = rsqrtf(red[0] / D + 1e-5f); __syncthreads();
  float rs = s_r;
  for (int c = t; c < D; c += 128) orow[c] = g[c] * (xr[c] - m) * rs + b[c];
}

__global__ void k_rbf(const float* __restrict__ ev, float* __restrict__ rbf) {
  long long i = (long long)blockIdx.x * 256 + threadIdx.x;
  if (i >= (long long)NEDGE * HIDC) return;
  int e = (int)(i >> 7), c = (int)(i & 127);
  float x = ev[e * 3], y = ev[e * 3 + 1], z = ev[e * 3 + 2];
  float d = sqrtf(x * x + y * y + z * z + 1e-12f);
  float center = 8.f * (float)c / 127.f;
  float gamma = 127.f / 8.f;
  float tt = d - center;
  rbf[i] = __expf(-gamma * tt * tt);
}

__global__ __launch_bounds__(256) void k_bn_stats(const float* __restrict__ x,
                                                  float* __restrict__ mean, float* __restrict__ var,
                                                  int M) {
  int c = blockIdx.x;   // channel 0..127
  float s = 0.f, sq = 0.f;
  for (int r = threadIdx.x; r < M; r += 256) {
    float v = x[(size_t)r * HIDC + c];
    s += v; sq += v * v;
  }
  __shared__ float rs[256], rq[256];
  rs[threadIdx.x] = s; rq[threadIdx.x] = sq; __syncthreads();
  for (int o = 128; o > 0; o >>= 1) {
    if (threadIdx.x < o) { rs[threadIdx.x] += rs[threadIdx.x + o]; rq[threadIdx.x] += rq[threadIdx.x + o]; }
    __syncthreads();
  }
  if (threadIdx.x == 0) {
    float m = rs[0] / M; mean[c] = m;
    float v = rq[0] / M - m * m; var[c] = v > 0.f ? v : 0.f;
  }
}

__global__ void k_bn_apply(const float* __restrict__ x, const float* __restrict__ g,
                           const float* __restrict__ b, const float* __restrict__ mean,
                           const float* __restrict__ var, const float* __restrict__ add,
                           float* __restrict__ out, long long n, int act) {
  long long i = (long long)blockIdx.x * 256 + threadIdx.x;
  if (i >= n) return;
  int c = (int)(i & (HIDC - 1));
  float y = g[c] * (x[i] - mean[c]) * rsqrtf(var[c] + 1e-5f) + b[c];
  y = actf(y, act);
  out[i] = add ? (y + add[i]) : y;
}

__global__ void k_build_z(const float* __restrict__ x, const float* __restrict__ est,
                          const int* __restrict__ srcI, const int* __restrict__ dstI,
                          float* __restrict__ z) {
  long long i = (long long)blockIdx.x * 256 + threadIdx.x;
  if (i >= (long long)NEDGE * 384) return;
  int e = (int)(i / 384), c = (int)(i % 384);
  float v;
  if (c < 128)      v = x[(size_t)dstI[e] * HIDC + c];
  else if (c < 256) v = x[(size_t)srcI[e] * HIDC + (c - 128)];
  else              v = est[(size_t)e * HIDC + (c - 256)];
  z[i] = v;
}

__global__ void k_edge_scatter(const float* __restrict__ sig, const float* __restrict__ msg,
                               const int* __restrict__ dstI, float* __restrict__ den,
                               float* __restrict__ agg) {
  long long i = (long long)blockIdx.x * 256 + threadIdx.x;
  if (i >= (long long)NEDGE * HIDC) return;
  int e = (int)(i >> 7), c = (int)(i & 127);
  float sg = sig[i];
  atomicAdd(&den[(size_t)dstI[e] * HIDC + c], sg);
  atomicAdd(&agg[(size_t)dstI[e] * HIDC + c], sg * msg[i]);
}

__global__ void k_div_eps(float* __restrict__ agg, const float* __restrict__ den, long long n) {
  long long i = (long long)blockIdx.x * 256 + threadIdx.x;
  if (i < n) agg[i] = agg[i] / (den[i] + 1e-6f);
}

// ---- GraphNorm (PyG): xc = y - a*mean[seg]; out = base + g*xc/sqrt(var[seg]+eps) + b ----
__global__ void k_gn_sum(const float* __restrict__ y, const int* __restrict__ seg,
                         float* __restrict__ gsum, float* __restrict__ cnt) {
  long long i = (long long)blockIdx.x * 256 + threadIdx.x;
  if (i >= (long long)NNODE * HIDC) return;
  int r = (int)(i >> 7), c = (int)(i & 127);
  int g = seg[r];
  atomicAdd(&gsum[g * HIDC + c], y[i]);
  if (c == 0) atomicAdd(&cnt[g], 1.f);
}
__global__ void k_gn_var(const float* __restrict__ y, const int* __restrict__ seg,
                         const float* __restrict__ a, const float* __restrict__ gsum,
                         const float* __restrict__ cnt, float* __restrict__ gvar) {
  long long i = (long long)blockIdx.x * 256 + threadIdx.x;
  if (i >= (long long)NNODE * HIDC) return;
  int r = (int)(i >> 7), c = (int)(i & 127);
  int g = seg[r];
  float cn = fmaxf(cnt[g], 1.f);
  float mean = gsum[g * HIDC + c] / cn;
  float xc = y[i] - a[c] * mean;
  atomicAdd(&gvar[g * HIDC + c], xc * xc);
}
__global__ void k_gn_apply(const float* __restrict__ y, const int* __restrict__ seg,
                           const float* __restrict__ a, const float* __restrict__ gg,
                           const float* __restrict__ gb, const float* __restrict__ gsum,
                           const float* __restrict__ cnt, const float* __restrict__ gvar,
                           const float* __restrict__ base, float* __restrict__ out) {
  long long i = (long long)blockIdx.x * 256 + threadIdx.x;
  if (i >= (long long)NNODE * HIDC) return;
  int r = (int)(i >> 7), c = (int)(i & 127);
  int g = seg[r];
  float cn = fmaxf(cnt[g], 1.f);
  float mean = gsum[g * HIDC + c] / cn;
  float xc = y[i] - a[c] * mean;
  float var = gvar[g * HIDC + c] / cn;
  out[i] = base[i] + gg[c] * xc * rsqrtf(var + 1e-5f) + gb[c];
}

// ---- Matformer helpers ----
__global__ void k_build_m(const float* __restrict__ v, const float* __restrict__ ef,
                          const int* __restrict__ srcI, const int* __restrict__ dstI,
                          float* __restrict__ m) {
  long long i = (long long)blockIdx.x * 256 + threadIdx.x;
  if (i >= (long long)EHROWS * 384) return;
  int row = (int)(i / 384), c = (int)(i % 384);
  int e = row >> 2, h = row & 3;
  float val;
  if (c < 128)      val = v[(size_t)dstI[e] * 512 + h * 128 + c];
  else if (c < 256) val = v[(size_t)srcI[e] * 512 + h * 128 + (c - 128)];
  else              val = ef[(size_t)e * 512 + h * 128 + (c - 256)];
  m[i] = val;
}

__global__ __launch_bounds__(128) void k_alpha_gate(const float* __restrict__ q,
    const float* __restrict__ k, const float* __restrict__ ef,
    const int* __restrict__ srcI, const int* __restrict__ dstI,
    const float* __restrict__ lng, const float* __restrict__ lnb,
    float* __restrict__ mbuf) {
  int row = blockIdx.x;          // e*HEADS + h
  int e = row >> 2, h = row & 3;
  int d = dstI[e], s = srcI[e];
  const float scale = 0.05103103630798288f;   // 1/sqrt(3*128)
  int t = threadIdx.x;
  float al[3];
#pragma unroll
  for (int jj = 0; jj < 3; ++jj) {
    int j = t + jj * 128;
    float qv = q[(size_t)d * 512 + h * 128 + (j & 127)];
    float kj;
    if (j < 128)      kj = k[(size_t)d * 512 + h * 128 + j];
    else if (j < 256) kj = k[(size_t)s * 512 + h * 128 + (j - 128)];
    else              kj = ef[(size_t)e * 512 + h * 128 + (j - 256)];
    al[jj] = qv * kj * scale;
  }
  __shared__ float red[128]; __shared__ float s_m, s_r;
  float sum = al[0] + al[1] + al[2];
  red[t] = sum; __syncthreads();
  for (int o = 64; o > 0; o >>= 1) { if (t < o) red[t] += red[t + o]; __syncthreads(); }
  if (t == 0) s_m = red[0] / 384.f; __syncthreads();
  float m = s_m, v = 0.f;
#pragma unroll
  for (int jj = 0; jj < 3; ++jj) { float dd = al[jj] - m; v += dd * dd; }
  red[t] = v; __syncthreads();
  for (int o = 64; o > 0; o >>= 1) { if (t < o) red[t] += red[t + o]; __syncthreads(); }
  if (t == 0) s_r = rsqrtf(red[0] / 384.f + 1e-5f); __syncthreads();
  float rs = s_r;
#pragma unroll
  for (int jj = 0; jj < 3; ++jj) {
    int j = t + jj * 128;
    float ln = lng[j] * (al[jj] - m) * rs + lnb[j];
    mbuf[(size_t)row * 384 + j] *= fast_sigmoid(ln);
  }
}

__global__ void k_head_scatter(const float* __restrict__ msg, const int* __restrict__ dstI,
                               float* __restrict__ out) {
  long long i = (long long)blockIdx.x * 256 + threadIdx.x;
  if (i >= (long long)EHROWS * HIDC) return;
  int row = (int)(i >> 7), c = (int)(i & 127);
  int e = row >> 2;
  atomicAdd(&out[(size_t)dstI[e] * HIDC + c], 0.25f * msg[i]);
}

__global__ void k_beta_build(const float* __restrict__ o, const float* __restrict__ xr,
                             float* __restrict__ out) {
  long long i = (long long)blockIdx.x * 256 + threadIdx.x;
  if (i >= (long long)NNODE * 384) return;
  int r = (int)(i / 384), c = (int)(i % 384);
  float v;
  if (c < 128)      v = o[(size_t)r * HIDC + c];
  else if (c < 256) v = xr[(size_t)r * HIDC + (c - 128)];
  else              v = o[(size_t)r * HIDC + (c - 256)] - xr[(size_t)r * HIDC + (c - 256)];
  out[i] = v;
}

__global__ __launch_bounds__(128) void k_rowdot(const float* __restrict__ A,
                                                const float* __restrict__ w,
                                                const float* __restrict__ b,
                                                float* __restrict__ out, int D, int act) {
  int row = blockIdx.x, t = threadIdx.x;
  float s = 0.f;
  for (int c = t; c < D; c += 128) s += A[(size_t)row * D + c] * w[c];
  __shared__ float red[128];
  red[t] = s; __syncthreads();
  for (int o = 64; o > 0; o >>= 1) { if (t < o) red[t] += red[t + o]; __syncthreads(); }
  if (t == 0) out[row] = actf(red[0] + b[0], act);
}

__global__ void k_hn(const float* __restrict__ bvec, const float* __restrict__ xr,
                     const float* __restrict__ o, float* __restrict__ hn) {
  long long i = (long long)blockIdx.x * 256 + threadIdx.x;
  if (i >= (long long)NNODE * HIDC) return;
  int r = (int)(i >> 7);
  float b = bvec[r];
  hn[i] = b * xr[i] + (1.f - b) * o[i];
}

__global__ void k_mix(const float* __restrict__ sA, const float* __restrict__ sB,
                      const float* __restrict__ hc, const float* __restrict__ hm,
                      float* __restrict__ hx) {
  long long i = (long long)blockIdx.x * 256 + threadIdx.x;
  if (i >= (long long)NNODE * HIDC) return;
  int r = (int)(i >> 7);
  float a = sA[r], b = sB[r];
  float mx = fmaxf(a, b);
  float ea = __expf(a - mx), eb = __expf(b - mx);
  float w0 = ea * __builtin_amdgcn_rcpf(ea + eb);
  hx[i] = w0 * hc[i] + (1.f - w0) * hm[i];
}

// ---- Set2Set ----
__global__ __launch_bounds__(128) void k_lstm(const float* __restrict__ qstar,
    const float* __restrict__ wih, const float* __restrict__ bih,
    const float* __restrict__ whh, const float* __restrict__ bhh,
    float* __restrict__ hh, float* __restrict__ cc) {
  int g = blockIdx.x, t = threadIdx.x;
  float acc[4];
#pragma unroll
  for (int j = 0; j < 4; ++j) {
    int col = t + j * 128;
    float s = bih[col] + bhh[col];
    for (int k2 = 0; k2 < 256; ++k2) s += qstar[g * 256 + k2] * wih[k2 * 512 + col];
    for (int k2 = 0; k2 < 128; ++k2) s += hh[g * 128 + k2] * whh[k2 * 512 + col];
    acc[j] = s;
  }
  __syncthreads();
  float ig = fast_sigmoid(acc[0]);
  float fg = fast_sigmoid(acc[1]);
  float gg = tanhf(acc[2]);
  float og = fast_sigmoid(acc[3]);
  float c = fg * cc[g * 128 + t] + ig * gg;
  cc[g * 128 + t] = c;
  hh[g * 128 + t] = og * tanhf(c);
}

__global__ void k_ener(const float* __restrict__ hx, const float* __restrict__ hh,
                       const int* __restrict__ seg, float* __restrict__ ener) {
  int n = blockIdx.x * blockDim.x + threadIdx.x;
  if (n >= NNODE) return;
  const float* hv = hh + (size_t)seg[n] * HIDC;
  const float* xv = hx + (size_t)n * HIDC;
  float s = 0.f;
  for (int c = 0; c < HIDC; ++c) s += xv[c] * hv[c];
  ener[n] = s;
}

__global__ __launch_bounds__(128) void k_s2s_attend(const float* __restrict__ ener,
    const float* __restrict__ hx, const float* __restrict__ hh,
    const int* __restrict__ seg, float* __restrict__ qstar) {
  int g = blockIdx.x, t = threadIdx.x;
  __shared__ float red[128]; __shared__ float s_mx, s_sum;
  float mx = -1e30f;
  for (int n = t; n < NNODE; n += 128) if (seg[n] == g) mx = fmaxf(mx, ener[n]);
  red[t] = mx; __syncthreads();
  for (int o = 64; o > 0; o >>= 1) { if (t < o) red[t] = fmaxf(red[t], red[t + o]); __syncthreads(); }
  if (t == 0) s_mx = (red[0] > -1e29f) ? red[0] : 0.f; __syncthreads();
  float sum = 0.f;
  for (int n = t; n < NNODE; n += 128) if (seg[n] == g) sum += __expf(ener[n] - s_mx);
  red[t] = sum; __syncthreads();
  for (int o = 64; o > 0; o >>= 1) { if (t < o) red[t] += red[t + o]; __syncthreads(); }
  if (t == 0) s_sum = red[0] + 1e-16f; __syncthreads();
  float rinv = __builtin_amdgcn_rcpf(s_sum);
  float r = 0.f;
  for (int n = 0; n < NNODE; ++n) {
    if (seg[n] == g) {
      float a = __expf(ener[n] - s_mx) * rinv;
      r += a * hx[(size_t)n * HIDC + t];
    }
  }
  qstar[g * 256 + t] = hh[g * HIDC + t];
  qstar[g * 256 + 128 + t] = r;
}

// =====================================================================
// host orchestration
// =====================================================================
static inline void gemm(hipStream_t s, const float* A, const float* Wt, const float* b,
                        float* C, int M, int N, int K, int act) {
  dim3 g((unsigned)(M / 128), (unsigned)(N / 128));
  if (act == 1)      k_gemm<1><<<g, dim3(256), 0, s>>>(A, Wt, b, C, M, N, K);
  else if (act == 2) k_gemm<2><<<g, dim3(256), 0, s>>>(A, Wt, b, C, M, N, K);
  else               k_gemm<0><<<g, dim3(256), 0, s>>>(A, Wt, b, C, M, N, K);
}
static inline dim3 ew(long long n) { return dim3((unsigned)((n + 255) / 256)); }
static inline void zero(hipStream_t s, float* p, long long n) {
  k_zero<<<ew(n), dim3(256), 0, s>>>(p, n);
}

extern "C" void kernel_launch(void* const* d_in, const int* in_sizes, int n_in,
                              void* d_out, int out_size, void* d_ws, size_t ws_size,
                              hipStream_t stream) {
  (void)in_sizes; (void)n_in; (void)out_size; (void)ws_size;
  const float* megnet  = (const float*)d_in[0];
  const float* edgevec = (const float*)d_in[1];
  const float* const* P = (const float* const*)(d_in + 2);   // 132 param leaves
  const int* atomn = (const int*)d_in[134];
  const int* eidx  = (const int*)d_in[135];
  const int* batch = (const int*)d_in[136];
  const int* srcI = eidx;            // edge_index[0]
  const int* dstI = eidx + NEDGE;    // edge_index[1]

  // param leaf indices (jax pytree flatten: dict keys sorted, lists in order)
  const float *ae_ln_b = P[0], *ae_ln_g = P[1], *ae_o1_b = P[2], *ae_o1_w = P[3],
              *ae_o2_b = P[4], *ae_o2_w = P[5];
  const int CART0 = 6;               // 15 leaves x 3 layers -> 6..50
  const float* embT = P[51];
  const float *fc1_b = P[52], *fc1_w = P[53], *fc2_b = P[54], *fc2_w = P[55],
              *gate_b = P[56], *gate_w = P[57];
  const int MAT0 = 58;               // 31 leaves x 2 layers -> 58..119
  const float *meg_b = P[120], *meg_w = P[121], *mix_b = P[122], *mix_w = P[123],
              *rbf1_b = P[124], *rbf1_w = P[125], *rbf2_b = P[126], *rbf2_w = P[127],
              *s2s_bhh = P[128], *s2s_bih = P[129], *s2s_whh = P[130], *s2s_wih = P[131];

  // ---------------- workspace carve ----------------
  float* ws = (float*)d_ws;
  size_t off = 0;
  auto AL = [&](size_t n) { float* p = ws + off; off += n; return p; };
  float* h0    = AL((size_t)NNODE * HIDC);
  float* x     = AL((size_t)NNODE * HIDC);      // CartNet state -> h_cart
  float* hM    = AL((size_t)NNODE * HIDC);      // Matformer state -> h_mat
  float* hx    = AL((size_t)NNODE * HIDC);
  float* efeat = AL((size_t)NEDGE * HIDC);
  float* est   = AL((size_t)NEDGE * HIDC);      // CartNet edge state
  float* qb    = AL((size_t)NNODE * 512);
  float* kb    = AL((size_t)NNODE * 512);
  float* vb    = AL((size_t)NNODE * 512);
  float* efb   = AL((size_t)NEDGE * 512);
  float* na    = AL((size_t)NNODE * 512);
  float* nb    = AL((size_t)NNODE * 512);
  float* nc    = AL((size_t)NNODE * 512);
  float* nd    = AL((size_t)NNODE * 512);
  float* den   = AL((size_t)NNODE * HIDC);
  float* agg   = AL((size_t)NNODE * HIDC);
  float* xl    = AL((size_t)NNODE * HIDC);
  float* outb  = AL((size_t)NNODE * HIDC);
  float* stm   = AL(512);
  float* stv   = AL(512);
  float* gsum  = AL((size_t)NGRAPH * HIDC);
  float* gvar  = AL((size_t)NGRAPH * HIDC);
  float* gcnt  = AL(NGRAPH);
  float* qstar = AL((size_t)NGRAPH * 256);
  float* hhb   = AL((size_t)NGRAPH * HIDC);
  float* ccb   = AL((size_t)NGRAPH * HIDC);
  float* enr   = AL(NNODE);
  float* sA    = AL(NNODE);
  float* sB    = AL(NNODE);
  float* bvec  = AL(NNODE);
  float* fc1o  = AL((size_t)NGRAPH * HIDC);
  float* bufA  = AL((size_t)EHROWS * 384);      // ~192 MB
  float* bufB  = AL((size_t)EHROWS * 384);      // ~192 MB

  const size_t NH = (size_t)NNODE * HIDC, EHID = (size_t)NEDGE * HIDC;

  // ---------------- AtomEncoder ----------------
  k_gather_rows<<<ew(NH), 256, 0, stream>>>(embT, atomn, na, NNODE, HIDC);       // emb
  k_lin_small<<<ew(NH), 256, 0, stream>>>(megnet, meg_w, meg_b, nb, NNODE, HIDC, 16, 0); // mg
  k_concat2<<<ew((long long)NNODE * 256), 256, 0, stream>>>(na, 128, nb, 128, nc, NNODE);
  gemm(stream, nc, gate_w, gate_b, nd, NNODE, HIDC, 256, 2);                     // sigmoid gate
  k_gate_fuse<<<ew(NH), 256, 0, stream>>>(nd, na, nb, nc);                       // fused
  gemm(stream, nc, ae_o1_w, ae_o1_b, nd, NNODE, HIDC, HIDC, 1);                  // silu(ae_o1)
  k_layernorm<<<NNODE, 128, 0, stream>>>(nd, ae_ln_g, ae_ln_b, na, HIDC);
  gemm(stream, na, ae_o2_w, ae_o2_b, h0, NNODE, HIDC, HIDC, 0);                  // h0

  // ---------------- RBF edge features ----------------
  k_rbf<<<ew(EHID), 256, 0, stream>>>(edgevec, bufA);
  gemm(stream, bufA, rbf1_w, rbf1_b, bufB, NEDGE, HIDC, HIDC, 1);
  gemm(stream, bufB, rbf2_w, rbf2_b, efeat, NEDGE, HIDC, HIDC, 0);               // efeat

  hipMemcpyAsync(x, h0, NH * 4, hipMemcpyDeviceToDevice, stream);
  hipMemcpyAsync(hM, h0, NH * 4, hipMemcpyDeviceToDevice, stream);
  hipMemcpyAsync(est, efeat, EHID * 4, hipMemcpyDeviceToDevice, stream);

  // ---------------- CartNet block x3 ----------------
  float* g1o = bufB;
  float* sig = bufB + ((size_t)8 << 20);
  float* a1o = bufB + ((size_t)16 << 20);
  float* a2o = bufB + ((size_t)24 << 20);
  for (int l = 0; l < 3; ++l) {
    int cb = CART0 + 15 * l;
    const float *a1_b = P[cb], *a1_w = P[cb + 1], *a2_b = P[cb + 2], *a2_w = P[cb + 3],
                *bne_b = P[cb + 4], *bne_g = P[cb + 5], *bnx_b = P[cb + 6], *bnx_g = P[cb + 7],
                *g1_b = P[cb + 8], *g1_w = P[cb + 9], *g2_b = P[cb + 10], *g2_w = P[cb + 11],
                *gn_a = P[cb + 12], *gn_b = P[cb + 13], *gn_g = P[cb + 14];
    k_build_z<<<ew((long long)NEDGE * 384), 256, 0, stream>>>(x, est, srcI, dstI, bufA);
    gemm(stream, bufA, g1_w, g1_b, g1o, NEDGE, HIDC, 384, 1);
    gemm(stream, g1o, g2_w, g2_b, est, NEDGE, HIDC, HIDC, 0);                    // e_ij -> edge state
    k_bn_stats<<<HIDC, 256, 0, stream>>>(est, stm, stv, NEDGE);
    k_bn_apply<<<ew(EHID), 256, 0, stream>>>(est, bne_g, bne_b, stm, stv, nullptr, sig, EHID, 2);
    gemm(stream, bufA, a1_w, a1_b, a1o, NEDGE, HIDC, 384, 1);
    gemm(stream, a1o, a2_w, a2_b, a2o, NEDGE, HIDC, HIDC, 0);
    zero(stream, den, NH); zero(stream, agg, NH);
    k_edge_scatter<<<ew(EHID), 256, 0, stream>>>(sig, a2o, dstI, den, agg);
    k_div_eps<<<ew(NH), 256, 0, stream>>>(agg, den, NH);
    k_bn_stats<<<HIDC, 256, 0, stream>>>(agg, stm, stv, NNODE);
    k_bn_apply<<<ew(NH), 256, 0, stream>>>(agg, bnx_g, bnx_b, stm, stv, x, xl, NH, 1); // x_l
    zero(stream, gsum, (size_t)NGRAPH * HIDC); zero(stream, gvar, (size_t)NGRAPH * HIDC);
    zero(stream, gcnt, NGRAPH);
    k_gn_sum<<<ew(NH), 256, 0, stream>>>(xl, batch, gsum, gcnt);
    k_gn_var<<<ew(NH), 256, 0, stream>>>(xl, batch, gn_a, gsum, gcnt, gvar);
    k_gn_apply<<<ew(NH), 256, 0, stream>>>(xl, batch, gn_a, gn_g, gn_b, gsum, gcnt, gvar, x, x);
  }

  // ---------------- Matformer block x2 ----------------
  for (int l = 0; l < 2; ++l) {
    int mb = MAT0 + 31 * l;
    const float *be_b = P[mb], *be_w = P[mb + 1], *bn_b = P[mb + 2], *bn_g = P[mb + 3],
                *f1_b = P[mb + 4], *f1_w = P[mb + 5], *f2_b = P[mb + 6], *f2_w = P[mb + 7],
                *gn_a = P[mb + 8], *gn_b = P[mb + 9], *gn_g = P[mb + 10],
                *lna_b = P[mb + 11], *lna_g = P[mb + 12],
                *ml_b = P[mb + 13], *ml_w = P[mb + 14],
                *mln_b = P[mb + 15], *mln_g = P[mb + 16],
                *mu1_b = P[mb + 17], *mu1_w = P[mb + 18],
                *mu2_b = P[mb + 19], *mu2_w = P[mb + 20],
                *sk_b = P[mb + 21], *sk_w = P[mb + 22],
                *we_b = P[mb + 23], *we_w = P[mb + 24],
                *wk_b = P[mb + 25], *wk_w = P[mb + 26],
                *wq_b = P[mb + 27], *wq_w = P[mb + 28],
                *wv_b = P[mb + 29], *wv_w = P[mb + 30];
    gemm(stream, hM, wq_w, wq_b, qb, NNODE, 512, HIDC, 0);
    gemm(stream, hM, wk_w, wk_b, kb, NNODE, 512, HIDC, 0);
    gemm(stream, hM, wv_w, wv_b, vb, NNODE, 512, HIDC, 0);
    gemm(stream, efeat, we_w, we_b, efb, NEDGE, 512, HIDC, 0);
    k_build_m<<<ew((long long)EHROWS * 384), 256, 0, stream>>>(vb, efb, srcI, dstI, bufA);
    gemm(stream, bufA, mu1_w, mu1_b, bufB, EHROWS, 384, 384, 1);                 // silu(mu1)
    gemm(stream, bufB, mu2_w, mu2_b, bufA, EHROWS, 384, 384, 0);                 // mu2
    k_alpha_gate<<<EHROWS, 128, 0, stream>>>(qb, kb, efb, srcI, dstI, lna_g, lna_b, bufA);
    gemm(stream, bufA, ml_w, ml_b, bufB, EHROWS, HIDC, 384, 0);                  // msg_lin
    k_layernorm<<<EHROWS, 128, 0, stream>>>(bufB, mln_g, mln_b, bufB, HIDC);     // msg_ln
    zero(stream, outb, NH);
    k_head_scatter<<<ew((long long)EHROWS * HIDC), 256, 0, stream>>>(bufB, dstI, outb);
    k_bn_stats<<<HIDC, 256, 0, stream>>>(outb, stm, stv, NNODE);
    k_bn_apply<<<ew(NH), 256, 0, stream>>>(outb, bn_g, bn_b, stm, stv, nullptr, outb, NH, 1);
    gemm(stream, hM, sk_w, sk_b, na, NNODE, HIDC, HIDC, 0);                      // xr
    k_beta_build<<<ew((long long)NNODE * 384), 256, 0, stream>>>(outb, na, bufA);
    k_rowdot<<<NNODE, 128, 0, stream>>>(bufA, be_w, be_b, bvec, 384, 2);         // sigmoid beta
    k_hn<<<ew(NH), 256, 0, stream>>>(bvec, na, outb, nc);
    gemm(stream, nc, f1_w, f1_b, nd, NNODE, 256, HIDC, 1);
    gemm(stream, nd, f2_w, f2_b, xl, NNODE, HIDC, 256, 0);                       // hn (FFN out)
    zero(stream, gsum, (size_t)NGRAPH * HIDC); zero(stream, gvar, (size_t)NGRAPH * HIDC);
    zero(stream, gcnt, NGRAPH);
    k_gn_sum<<<ew(NH), 256, 0, stream>>>(xl, batch, gsum, gcnt);
    k_gn_var<<<ew(NH), 256, 0, stream>>>(xl, batch, gn_a, gsum, gcnt, gvar);
    k_gn_apply<<<ew(NH), 256, 0, stream>>>(xl, batch, gn_a, gn_g, gn_b, gsum, gcnt, gvar, hM, hM);
  }

  // ---------------- AttentionFusionMixer ----------------
  k_rowdot<<<NNODE, 128, 0, stream>>>(x, mix_w, mix_b, sA, HIDC, 0);
  k_rowdot<<<NNODE, 128, 0, stream>>>(hM, mix_w, mix_b, sB, HIDC, 0);
  k_mix<<<ew(NH), 256, 0, stream>>>(sA, sB, x, hM, hx);

  // ---------------- Set2Set (3 steps) ----------------
  zero(stream, qstar, (size_t)NGRAPH * 256);
  zero(stream, hhb, (size_t)NGRAPH * HIDC);
  zero(stream, ccb, (size_t)NGRAPH * HIDC);
  for (int it = 0; it < 3; ++it) {
    k_lstm<<<NGRAPH, 128, 0, stream>>>(qstar, s2s_wih, s2s_bih, s2s_whh, s2s_bhh, hhb, ccb);
    k_ener<<<(NNODE + 255) / 256, 256, 0, stream>>>(hx, hhb, batch, enr);
    k_s2s_attend<<<NGRAPH, 128, 0, stream>>>(enr, hx, hhb, batch, qstar);
  }

  // ---------------- readout MLP ----------------
  gemm(stream, qstar, fc1_w, fc1_b, fc1o, NGRAPH, HIDC, 256, 1);
  k_rowdot<<<NGRAPH, 128, 0, stream>>>(fc1o, fc2_w, fc2_b, (float*)d_out, HIDC, 0);
}
